// MeanNetAggregator_37168646979928
// MI455X (gfx1250) — matrix operand order repaired and checked by reading, not compiled, not run on hardware
//
#include <hip/hip_runtime.h>
#include <hip/hip_bf16.h>

typedef __attribute__((ext_vector_type(16))) _Float16     v16h;
typedef __attribute__((ext_vector_type(8)))  float        v8f;
typedef __attribute__((ext_vector_type(8)))  unsigned int v8u;

#define NUM_NODES 16384
#define NUM_NETS  8192
#define NODE_DIM  256
#define NKT       (NUM_NODES / 32)   // 512 K-tiles of 32
#define ROWS_PER_BLOCK 32            // 2 WMMA M-tiles per block

// ---- helpers -------------------------------------------------------------

// Two mask bytes (values 0/1) -> packed f16 pair {b1*1.0h, b0*1.0h} in one
// v_perm_b32 (bytes to bit 0 / bit 16) + one v_mul (x * 0x3C00).
__device__ __forceinline__ unsigned int pair_lo(unsigned int x) {
  return __builtin_amdgcn_perm(0u, x, 0x0C010C00u) * 0x3C00u; // bytes 0,1
}
__device__ __forceinline__ unsigned int pair_hi(unsigned int x) {
  return __builtin_amdgcn_perm(0u, x, 0x0C030C02u) * 0x3C00u; // bytes 2,3
}

// Convert 16 mask bytes (two b64 loads) into the 16x32 f16 A-operand layout:
// this lane's halves 0..7 = first 8 bytes, halves 8..15 = second 8 bytes.
__device__ __forceinline__ v16h cvt_mask16(unsigned long long a0,
                                           unsigned long long a1) {
  v8u w;
  const unsigned int x0 = (unsigned int)a0, x1 = (unsigned int)(a0 >> 32);
  const unsigned int y0 = (unsigned int)a1, y1 = (unsigned int)(a1 >> 32);
  w[0] = pair_lo(x0); w[1] = pair_hi(x0);
  w[2] = pair_lo(x1); w[3] = pair_hi(x1);
  w[4] = pair_lo(y0); w[5] = pair_hi(y0);
  w[6] = pair_lo(y1); w[7] = pair_hi(y1);
  return __builtin_bit_cast(v16h, w);
}

// ---- prep: split f32 embeddings into f16 hi/lo, pre-swizzled into the
// WMMA B-operand (32x16 f16) register layout so the main loop does pure
// contiguous b128 loads. One thread per (even-k pair, dim).
// B element (k,n) of tile (kt=k/32, nt=n/16):
//   lane = (n&15) + 16*((k>>3)&1), half index = (k&7) + 8*((k>>4)&1) (2 halves per VGPR).
__global__ __launch_bounds__(256)
void pack_b_hilo(const float* __restrict__ E,
                 unsigned int* __restrict__ bhi,
                 unsigned int* __restrict__ blo) {
  const int tid = blockIdx.x * 256 + threadIdx.x;
  const int n = tid & (NODE_DIM - 1);
  const int k = (tid >> 8) << 1;                  // even k
  const float e0 = E[(size_t)k * NODE_DIM + n];
  const float e1 = E[(size_t)(k + 1) * NODE_DIM + n];
  const _Float16 h0 = (_Float16)e0;
  const _Float16 l0 = (_Float16)(e0 - (float)h0); // residual split (hi+lo ~ f32)
  const _Float16 h1 = (_Float16)e1;
  const _Float16 l1 = (_Float16)(e1 - (float)h1);
  const int kt = k >> 5;
  const int nt = n >> 4;
  const int lane = (n & 15) | ((k & 8) << 1);
  const int v = ((k & 7) >> 1) | ((k & 16) >> 2); // u32 slot within lane
  const size_t idx = ((size_t)(kt * 16 + nt) * 32 + lane) * 8 + v;
  bhi[idx] = (unsigned int)__builtin_bit_cast(unsigned short, h0) |
             ((unsigned int)__builtin_bit_cast(unsigned short, h1) << 16);
  blo[idx] = (unsigned int)__builtin_bit_cast(unsigned short, l0) |
             ((unsigned int)__builtin_bit_cast(unsigned short, l1) << 16);
}

// ---- main: 8 waves/block; block = 32 net-rows x full 256 dims.
// wave w: M-tiles {0,1}, N-tiles {2w, 2w+1}; 8 v_wmma per K-step of 32.
__global__ __launch_bounds__(256)
void net_mean_wmma(const unsigned char* __restrict__ mask,
                   const v16h* __restrict__ Bhi,
                   const v16h* __restrict__ Blo,
                   float* __restrict__ out) {
  __shared__ float sRecip[ROWS_PER_BLOCK];
  const int lane = threadIdx.x & 31;
  const int w = threadIdx.x >> 5;
  const int r0 = blockIdx.x * ROWS_PER_BLOCK;
  const int mrow = lane & 15;
  const int khi = (lane >> 4) << 3;               // 0 or 8: A K-offset per lane group

  const unsigned char* a0p = mask + (size_t)(r0 + mrow) * NUM_NODES + khi;
  const unsigned char* a1p = a0p + (size_t)16 * NUM_NODES;

  const size_t bBase0 = (size_t)(2 * w) * 32 + lane;
  const size_t bBase1 = (size_t)(2 * w + 1) * 32 + lane;

  v8f c00 = {}, c01 = {}, c10 = {}, c11 = {};
  unsigned int cnt = 0;
  const bool countHiTile = (w & 1) != 0;          // which 16-row tile this wave counts

  for (int ktO = 0; ktO < NKT; ktO += 4) {
    // byte-lane accumulator: per-byte sum <= 2 per step, <= 8 over 4 steps,
    // total <= 64 so the *0x0101..>>56 reduction below cannot overflow.
    unsigned long long acc = 0;
#pragma unroll
    for (int kti = 0; kti < 4; ++kti) {
      const int ko = (ktO + kti) << 5;
      // mask bytes: streamed once, non-temporal so they don't evict B from L2
      const unsigned long long a00 =
          __builtin_nontemporal_load((const unsigned long long*)(a0p + ko));
      const unsigned long long a01 =
          __builtin_nontemporal_load((const unsigned long long*)(a0p + ko + 16));
      const unsigned long long a10 =
          __builtin_nontemporal_load((const unsigned long long*)(a1p + ko));
      const unsigned long long a11 =
          __builtin_nontemporal_load((const unsigned long long*)(a1p + ko + 16));

      const size_t bo = (size_t)(ktO + kti) * (16 * 32);
      const v16h bh0 = Bhi[bo + bBase0];
      const v16h bh1 = Bhi[bo + bBase1];
      const v16h bl0 = Blo[bo + bBase0];
      const v16h bl1 = Blo[bo + bBase1];

      const v16h am0 = cvt_mask16(a00, a01);
      const v16h am1 = cvt_mask16(a10, a11);

      acc += countHiTile ? (a10 + a11) : (a00 + a01);

      // hi pass then lo pass; 4 independent accumulators between reuses
      c00 = __builtin_amdgcn_wmma_f32_16x16x32_f16(false, am0, false, bh0, (short)0, c00, false, false);
      c01 = __builtin_amdgcn_wmma_f32_16x16x32_f16(false, am0, false, bh1, (short)0, c01, false, false);
      c10 = __builtin_amdgcn_wmma_f32_16x16x32_f16(false, am1, false, bh0, (short)0, c10, false, false);
      c11 = __builtin_amdgcn_wmma_f32_16x16x32_f16(false, am1, false, bh1, (short)0, c11, false, false);
      c00 = __builtin_amdgcn_wmma_f32_16x16x32_f16(false, am0, false, bl0, (short)0, c00, false, false);
      c01 = __builtin_amdgcn_wmma_f32_16x16x32_f16(false, am0, false, bl1, (short)0, c01, false, false);
      c10 = __builtin_amdgcn_wmma_f32_16x16x32_f16(false, am1, false, bl0, (short)0, c10, false, false);
      c11 = __builtin_amdgcn_wmma_f32_16x16x32_f16(false, am1, false, bl1, (short)0, c11, false, false);
    }
    // fold 8 byte-lane partial sums (each <= 8, total <= 64)
    cnt += (unsigned int)((acc * 0x0101010101010101ull) >> 56);
  }

  // lane L holds K-subset {0-7,16-23,...}, lane L+16 the complement:
  // fold with SWAPX16 swizzle (xor 0x10) so lanes 0-15 hold full row counts.
  cnt += (unsigned int)__builtin_amdgcn_ds_swizzle((int)cnt, 0x401F);
  if (w < 2 && lane < 16) {
    sRecip[w * 16 + lane] = (cnt > 0) ? (1.0f / (float)cnt) : 0.0f;
  }
  __syncthreads();

  // C/D tile layout: lane 0-15 -> N=lane, M=r;  lane 16-31 -> N=lane-16, M=r+8
  const int rsub = (lane >> 4) << 3;
  const int col0 = w * 32 + (lane & 15);
#pragma unroll
  for (int r = 0; r < 8; ++r) {
    const float s0 = sRecip[r + rsub];
    const float s1 = sRecip[16 + r + rsub];
    out[(size_t)(r0 + r + rsub) * NODE_DIM + col0]            = c00[r] * s0;
    out[(size_t)(r0 + r + rsub) * NODE_DIM + col0 + 16]       = c01[r] * s0;
    out[(size_t)(r0 + 16 + r + rsub) * NODE_DIM + col0]       = c10[r] * s1;
    out[(size_t)(r0 + 16 + r + rsub) * NODE_DIM + col0 + 16]  = c11[r] * s1;
  }
}

// ---- launch --------------------------------------------------------------
extern "C" void kernel_launch(void* const* d_in, const int* in_sizes, int n_in,
                              void* d_out, int out_size, void* d_ws, size_t ws_size,
                              hipStream_t stream) {
  const float* E = (const float*)d_in[0];                      // [16384,256] f32
  const unsigned char* mask = (const unsigned char*)d_in[1];   // [8192,16384] bool (1 byte/elem)
  float* out = (float*)d_out;                                  // [8192,256] f32

  // workspace: 8 MB f16-hi + 8 MB f16-lo, pre-swizzled B operands
  unsigned int* bhi = (unsigned int*)d_ws;
  unsigned int* blo = bhi + (size_t)NUM_NODES * NODE_DIM / 2;  // u32 elements

  pack_b_hilo<<<(NUM_NODES / 2) * NODE_DIM / 256, 256, 0, stream>>>(E, bhi, blo);
  net_mean_wmma<<<NUM_NETS / ROWS_PER_BLOCK, 256, 0, stream>>>(
      mask, (const v16h*)bhi, (const v16h*)blo, out);
}
